// EquiMessagePassingLayer_23871428231783
// MI455X (gfx1250) — compile-verified, matrix operand order checked by reading, not run on hardware
//
#include <hip/hip_runtime.h>
#include <hip/hip_bf16.h>
#include <math.h>

// ---------------------------------------------------------------------------
// Problem constants
// ---------------------------------------------------------------------------
#define BB 8
#define SS 64
#define NN 256
#define DM 384
#define DMSG 64
#define HH 32
#define DH 12
#define DOUT 96

typedef _Float16 half_t;
typedef __attribute__((ext_vector_type(16))) _Float16 v16h;
typedef __attribute__((ext_vector_type(8)))  float    v8f;

// ---------------------------------------------------------------------------
// Workspace layout (byte offsets)
// ---------------------------------------------------------------------------
constexpr size_t SZ_C    = (size_t)BB*SS*NN*3*4;
constexpr size_t OFF_C   = 0;
constexpr size_t OFF_CN  = OFF_C + SZ_C;
constexpr size_t OFF_PROJ= OFF_CN + SZ_C;
constexpr size_t OFF_HA  = OFF_PROJ + (size_t)BB*3*NN*64*4;
constexpr size_t OFF_HB  = OFF_HA + (size_t)BB*NN*96*4;
constexpr size_t OFF_LNA = OFF_HB + (size_t)BB*NN*96*4;
constexpr size_t OFF_Q   = OFF_LNA + (size_t)BB*NN*DM*2;
constexpr size_t OFF_OUTH= OFF_Q + (size_t)BB*NN*DM*4;
constexpr size_t OFF_UPD = OFF_OUTH + (size_t)BB*NN*DM*2;
constexpr size_t OFF_RD  = OFF_UPD + (size_t)BB*3*NN*64*4;
constexpr size_t OFF_W1C = OFF_RD + (size_t)BB*NN*4;
constexpr size_t OFF_W2T = OFF_W1C + (size_t)64*96*2;
constexpr size_t OFF_INT = OFF_W2T + (size_t)96*96*2;
constexpr size_t OFF_OUTT= OFF_INT + (size_t)DM*DM*2;

// ---------------------------------------------------------------------------
// WMMA 16-bit operand layout (CDNA5 ISA 7.12.2). For lane L (hi = L>=16),
// v16h element e holds K index:
//   kmap(e,hi) = (e&7) + (hi ? 8 : 0) + ((e&8)<<1)
// Inverse (given kr in [0,32)): hi = (kr>>3)&1 ; e = (kr&7) | ((kr>>1)&8)
// B operand assumed symmetric (lane -> N column, same K packing).
// ---------------------------------------------------------------------------
__device__ __host__ __forceinline__ int kmap(int e, int hi) {
    return (e & 7) + (hi ? 8 : 0) + ((e & 8) << 1);
}

// Fragment stored pre-packed: frag base + lane*16 halves, contiguous 32 bytes.
__device__ __forceinline__ v16h load_packed(const half_t* p, int lane) {
    union { v16h v; uint4 q[2]; } f;
    const uint4* src = (const uint4*)(p + lane * 16);
    f.q[0] = src[0]; f.q[1] = src[1];
    return f.v;
}
// A fragment from a row-major LDS tile using paired (K,K+1) 32-bit loads.
__device__ __forceinline__ v16h load_a_pairs(const half_t* base, int stride, int lane) {
    union { v16h v; unsigned u[8]; } f;
    const int m = lane & 15, hi = lane >> 4;
#pragma unroll
    for (int e2 = 0; e2 < 8; ++e2) {
        int k = kmap(2 * e2, hi);
        f.u[e2] = *(const unsigned*)&base[m * stride + k];
    }
    return f.v;
}
__device__ __forceinline__ v8f wmma32(v16h a, v16h b, v8f c) {
    return __builtin_amdgcn_wmma_f32_16x16x32_f16(false, a, false, b, (short)0, c, false, false);
}

// ---------------------------------------------------------------------------
// K0: convert weights to f16, pre-packed in WMMA B-fragment order:
//   buf[((kc*NT + nt)*32 + lane)*16 + e] = W[nt*16 + (lane&15)][kc*32 + kmap(e,lane>=16)]
// ---------------------------------------------------------------------------
__global__ void k_prep_weights(const float* __restrict__ w1, const float* __restrict__ w2,
                               const float* __restrict__ in_w, const float* __restrict__ out_w,
                               half_t* w1cP, half_t* w2P, half_t* inP, half_t* outP) {
    const int T0 = 64*96, T1 = T0 + 96*96, T2 = T1 + DM*DM, T3 = T2 + DM*DM;
    for (int i = blockIdx.x * blockDim.x + threadIdx.x; i < T3; i += gridDim.x * blockDim.x) {
        if (i < T0) {                       // w1c: f x s  (96 x 64), NT=6, KC=2
            int e = i & 15, lane = (i >> 4) & 31, rest = i >> 9;
            int nt = rest % 6, kc = rest / 6;
            int f = nt*16 + (lane & 15);
            int s = kc*32 + kmap(e, lane >> 4);
            w1cP[i] = (half_t)w1[f*192 + 128 + s];
        } else if (i < T1) {                // w2: f x k (96 x 96), NT=6, KC=3
            int j = i - T0;
            int e = j & 15, lane = (j >> 4) & 31, rest = j >> 9;
            int nt = rest % 6, kc = rest / 6;
            int f = nt*16 + (lane & 15);
            int k = kc*32 + kmap(e, lane >> 4);
            w2P[j] = (half_t)w2[f*96 + k];
        } else if (i < T2) {                // in_w: n x k (384 x 384), NT=24, KC=12
            int j = i - T1;
            int e = j & 15, lane = (j >> 4) & 31, rest = j >> 9;
            int nt = rest % 24, kc = rest / 24;
            int n = nt*16 + (lane & 15);
            int k = kc*32 + kmap(e, lane >> 4);
            inP[j] = (half_t)in_w[n*DM + k];
        } else {                            // out_w
            int j = i - T2;
            int e = j & 15, lane = (j >> 4) & 31, rest = j >> 9;
            int nt = rest % 24, kc = rest / 24;
            int n = nt*16 + (lane & 15);
            int k = kc*32 + kmap(e, lane >> 4);
            outP[j] = (half_t)out_w[n*DM + k];
        }
    }
}

// ---------------------------------------------------------------------------
// K1: per-(b,s) coordinate normalization -> c (norm 'none') and cn ('length')
// ---------------------------------------------------------------------------
__global__ void k_coord_prep(const float* __restrict__ coords, const float* __restrict__ mask,
                             const float* __restrict__ w_em, const float* __restrict__ w_ca,
                             float* __restrict__ c_out, float* __restrict__ cn_out) {
    const int bs = blockIdx.x;              // b*SS + s
    const int s  = bs % SS;
    const int n  = threadIdx.x;             // 256
    __shared__ float red[256];
    const int base = (bs * NN + n) * 3;
    const float m = mask[bs * NN + n];
    const float x = coords[base], y = coords[base+1], z = coords[base+2];

    auto blk = [&](float v) -> float {
        red[n] = v; __syncthreads();
        for (int o = 128; o > 0; o >>= 1) { if (n < o) red[n] += red[n+o]; __syncthreads(); }
        float r = red[0]; __syncthreads(); return r;
    };
    const float sm_ = blk(m);
    const float sx = blk(m*x), sy = blk(m*y), sz = blk(m*z);
    const float cx = (x - sx/sm_) * m, cy = (y - sy/sm_) * m, cz = (z - sz/sm_) * m;
    const float we = w_em[s], wc = w_ca[s];
    c_out[base] = cx*we; c_out[base+1] = cy*we; c_out[base+2] = cz*we;
    const float len = sqrtf(cx*cx + cy*cy + cz*cz + 1e-12f);
    const float sl = blk(len);
    const float sc = wc / (sl/sm_ + 1e-6f);
    cn_out[base] = cx*sc; cn_out[base+1] = cy*sc; cn_out[base+2] = cz*sc;
}

// ---------------------------------------------------------------------------
// K1b: proj[b,d,n,p] = sum_s cn[b,s,n,d] * ca_proj_w[p,s]
// ---------------------------------------------------------------------------
__global__ void k_proj(const float* __restrict__ cn, const float* __restrict__ pw,
                       float* __restrict__ proj) {
    const int bn = blockIdx.x;  const int b = bn >> 8, n = bn & 255;
    __shared__ float W[64][64];
    __shared__ float cl[64][3];
    const int t = threadIdx.x;  // 192
    for (int i = t; i < 4096; i += 192) W[i >> 6][i & 63] = pw[i];
    { int s = t / 3, d = t % 3; cl[s][d] = cn[((b*SS + s)*NN + n)*3 + d]; }
    __syncthreads();
    const int d = t / 64, p = t % 64;
    float acc = 0.f;
#pragma unroll 8
    for (int s = 0; s < 64; ++s) acc += W[p][s] * cl[s][d];
    proj[((b*3 + d)*NN + n)*64 + p] = acc;
}

// ---------------------------------------------------------------------------
// K2: per-node LN, h = LNe@np_w.T+np_b, hA/hB precontraction, LNa (f16), degree
// ---------------------------------------------------------------------------
__global__ void k_node_prep(const float* __restrict__ nf,
                            const float* __restrict__ em_g, const float* __restrict__ em_b,
                            const float* __restrict__ na_g, const float* __restrict__ na_b,
                            const float* __restrict__ np_w, const float* __restrict__ np_b,
                            const float* __restrict__ w1, const float* __restrict__ b1,
                            const int* __restrict__ adj,
                            half_t* __restrict__ LNa, float* __restrict__ hA,
                            float* __restrict__ hB, int* __restrict__ rowdeg) {
    const int bn = blockIdx.x;
    const float* x = nf + bn * DM;
    __shared__ float xs[DM], xhat[DM], hm[64], red[128];
    const int t = threadIdx.x;  // 128
    float ls = 0.f;
    for (int k = t; k < DM; k += 128) { float v = x[k]; xs[k] = v; ls += v; }
    red[t] = ls; __syncthreads();
    for (int o = 64; o > 0; o >>= 1) { if (t < o) red[t] += red[t+o]; __syncthreads(); }
    const float mu = red[0] / (float)DM; __syncthreads();
    float lv = 0.f;
    for (int k = t; k < DM; k += 128) { float d = xs[k] - mu; lv += d*d; }
    red[t] = lv; __syncthreads();
    for (int o = 64; o > 0; o >>= 1) { if (t < o) red[t] += red[t+o]; __syncthreads(); }
    const float rstd = rsqrtf(red[0] / (float)DM + 1e-5f); __syncthreads();
    for (int k = t; k < DM; k += 128) {
        float hh = (xs[k] - mu) * rstd; xhat[k] = hh;
        LNa[bn*DM + k] = (half_t)(hh * na_g[k] + na_b[k]);
    }
    __syncthreads();
    if (t < 64) {
        float s = np_b[t];
        for (int k = 0; k < DM; ++k) s += (xhat[k]*em_g[k] + em_b[k]) * np_w[t*DM + k];
        hm[t] = s;
    }
    __syncthreads();
    if (t < 96) {
        float sa = 0.f, sb = b1[t];
        for (int k = 0; k < 64; ++k) { float h = hm[k]; sa += h * w1[t*192 + k]; sb += h * w1[t*192 + 64 + k]; }
        hA[bn*96 + t] = sa; hB[bn*96 + t] = sb;
    }
    int rs = 0;
    for (int j = t; j < NN; j += 128) rs += adj[bn*NN + j];
    __syncthreads();
    red[t] = (float)rs; __syncthreads();
    for (int o = 64; o > 0; o >>= 1) { if (t < o) red[t] += red[t+o]; __syncthreads(); }
    if (t == 0) rowdeg[bn] = (int)red[0];
}

// ---------------------------------------------------------------------------
// Generic WMMA GEMM: C[M,Nn] = A[M,K](f16) @ W(pre-packed frag f16) + bias (+resid)
// Block tile 128x64, 8 waves (4 x 2); B fragments straight from global/L2.
// All hot indices are 32-bit so codegen can use saddr+voffset addressing.
// ---------------------------------------------------------------------------
__global__ void k_gemm(const half_t* __restrict__ A, const half_t* __restrict__ Bp,
                       const float* __restrict__ bias, const float* __restrict__ resid,
                       float* __restrict__ C, int M, int K, int Nn) {
    __shared__ half_t At[128][40];
    const int t = threadIdx.x, lane = t & 31, w = t >> 5;
    const int lane15 = lane & 15, hi = lane >> 4;
    const int wm = w & 3, wn = w >> 2;
    const int row0 = blockIdx.x * 128, col0 = blockIdx.y * 64;
    const int ntiles = Nn >> 4;
    const int nt0 = (col0 >> 4) + wn * 2;
    v8f vz = {0.f,0.f,0.f,0.f,0.f,0.f,0.f,0.f};
    v8f acc[2][2] = {{vz, vz}, {vz, vz}};
    for (int kk = 0, kc = 0; kk < K; kk += 32, ++kc) {
        for (int i = t; i < 512; i += 256) {       // 128 rows x 32 halves, uint4 chunks
            int r = i >> 2, c4 = i & 3;
            *(uint4*)&At[r][c4 * 8] = *(const uint4*)&A[(row0 + r)*K + kk + c4*8];
        }
        __syncthreads();
        v16h b0 = load_packed(Bp + (kc * ntiles + nt0) * 512, lane);
        v16h b1 = load_packed(Bp + (kc * ntiles + nt0 + 1) * 512, lane);
        v16h a0 = load_a_pairs(&At[wm*32][0], 40, lane);
        v16h a1 = load_a_pairs(&At[wm*32 + 16][0], 40, lane);
        acc[0][0] = wmma32(a0, b0, acc[0][0]);
        acc[0][1] = wmma32(a0, b1, acc[0][1]);
        acc[1][0] = wmma32(a1, b0, acc[1][0]);
        acc[1][1] = wmma32(a1, b1, acc[1][1]);
        __syncthreads();
    }
#pragma unroll
    for (int ms = 0; ms < 2; ++ms)
#pragma unroll
        for (int ns = 0; ns < 2; ++ns)
#pragma unroll
            for (int r = 0; r < 8; ++r) {
                int row = row0 + wm*32 + ms*16 + r + hi*8;
                int col = col0 + wn*32 + ns*16 + lane15;
                float v = acc[ms][ns][r] + bias[col];
                if (resid) v += resid[row*Nn + col];
                C[row*Nn + col] = v;
            }
}

// ---------------------------------------------------------------------------
// Fused message MLP + dual online-softmax attention kernel.
// Grid: BB*16 workgroups, one per (b, 16-row i-tile). 256 threads, 8 waves.
// cf and silu activations are written to LDS directly in A-fragment packed
// order, so every WMMA operand read is 2 x ds_load_b128.
// ---------------------------------------------------------------------------
struct FusedLDS {
    // persistent across the j-loop
    float ci[64][16][3];
    float proji[16][64][3];
    float hAi[16][96];
    float nstate[16][32][3];    // m, l, ssq
    float nacc[16][32][12];
    float cstate[16][64][3];
    float cacc[16][64][3];
    int   adjt[16][16];
    int   rowdeg[16];
    float b2s[96];
    union {
        struct {                        // GEMM stages
            float  cj[64][16][3];       // 12288 B
            half_t cfP[16*2*512];       // 32768 B  (chunk,kc) packed A frags
            half_t xP[16*3*512];        // 49152 B
            float  hBj[16][96];         // 6144 B
        } g;
        struct {                        // softmax / accumulation stage
            float  nodeP[16][16][32];
            float  coordP[16][16][32];
            half_t qj[16][DM];
            float  projj[16][64][3];
            float  nfac[16][32];
            float  cfac[16][64];
        } a;
    } u;
};

__global__ void k_fused(const float* __restrict__ c_ws, const float* __restrict__ proj_ws,
                        const float* __restrict__ hA_ws, const float* __restrict__ hB_ws,
                        const float* __restrict__ q_ws, const int* __restrict__ rd_ws,
                        const int* __restrict__ adj,
                        const half_t* __restrict__ w1cP, const half_t* __restrict__ w2P,
                        const float* __restrict__ b2,
                        half_t* __restrict__ out_h, float* __restrict__ upd_ws) {
    __shared__ FusedLDS sm;
    const int tid = threadIdx.x, lane = tid & 31, w = tid >> 5;
    const int lane15 = lane & 15, hi = lane >> 4;
    const int b = blockIdx.x >> 4;
    const int i0 = (blockIdx.x & 15) * 16;

    // ---- persistent loads & state init ----
    for (int idx = tid; idx < 64*16*3; idx += 256) {
        int s = idx / 48, r = idx % 48, ii = r / 3, d = r % 3;
        sm.ci[s][ii][d] = c_ws[((b*SS + s)*NN + i0 + ii)*3 + d];
    }
    for (int idx = tid; idx < 16*64*3; idx += 256) {
        int ii = idx / 192, r = idx % 192, p = r / 3, d = r % 3;
        sm.proji[ii][p][d] = proj_ws[((b*3 + d)*NN + i0 + ii)*64 + p];
    }
    for (int idx = tid; idx < 16*96; idx += 256) {
        int ii = idx / 96, f = idx % 96;
        sm.hAi[ii][f] = hA_ws[(b*NN + i0 + ii)*96 + f];
    }
    for (int idx = tid; idx < 16*32; idx += 256) {
        sm.nstate[idx/32][idx%32][0] = -1e30f;
        sm.nstate[idx/32][idx%32][1] = 0.f;
        sm.nstate[idx/32][idx%32][2] = 0.f;
    }
    for (int idx = tid; idx < 16*64; idx += 256) {
        sm.cstate[idx/64][idx%64][0] = -1e30f;
        sm.cstate[idx/64][idx%64][1] = 0.f;
        sm.cstate[idx/64][idx%64][2] = 0.f;
    }
    for (int idx = tid; idx < 16*32*12; idx += 256) ((float*)sm.nacc)[idx] = 0.f;
    for (int idx = tid; idx < 16*64*3;  idx += 256) ((float*)sm.cacc)[idx] = 0.f;
    if (tid < 96) sm.b2s[tid] = b2[tid];
    if (tid < 16) sm.rowdeg[tid] = rd_ws[b*NN + i0 + tid];
    __syncthreads();

    v8f acc[2][6];
    const v8f vz = {0.f,0.f,0.f,0.f,0.f,0.f,0.f,0.f};

    for (int jt = 0; jt < 16; ++jt) {
        const int j0 = jt * 16;
        // ---- stage 1: load cj, adj tile, hBj ----
        for (int idx = tid; idx < 64*16*3; idx += 256) {
            int s = idx / 48, r = idx % 48, jj = r / 3, d = r % 3;
            sm.u.g.cj[s][jj][d] = c_ws[((b*SS + s)*NN + j0 + jj)*3 + d];
        }
        {
            int ii = tid >> 4, jj = tid & 15;
            sm.adjt[ii][jj] = adj[(b*NN + i0 + ii)*NN + j0 + jj];
        }
        for (int idx = tid; idx < 16*96; idx += 256) {
            int jj = idx / 96, f = idx % 96;
            sm.u.g.hBj[jj][f] = hB_ws[(b*NN + j0 + jj)*96 + f];
        }
        __syncthreads();
        // ---- coord_feats tile -> cfP (packed A-frag order) ----
        {
            const int ii = tid >> 4, jj = tid & 15;
#pragma unroll 8
            for (int s = 0; s < 64; ++s) {
                float d0 = sm.ci[s][ii][0]*sm.u.g.cj[s][jj][0]
                         + sm.ci[s][ii][1]*sm.u.g.cj[s][jj][1]
                         + sm.ci[s][ii][2]*sm.u.g.cj[s][jj][2];
                int kc = s >> 5, kr = s & 31;
                int h2 = (kr >> 3) & 1;
                int e  = (kr & 7) | ((kr >> 1) & 8);
                sm.u.g.cfP[((ii*2 + kc)*32 + jj + h2*16)*16 + e] = (half_t)d0;
            }
        }
        __syncthreads();
        // ---- GEMM1: pre = cf @ W1c^T (+hA+hB), silu -> xP (packed) ----
#pragma unroll
        for (int mi = 0; mi < 2; ++mi)
#pragma unroll
            for (int nt = 0; nt < 6; ++nt) acc[mi][nt] = vz;
#pragma unroll
        for (int kc = 0; kc < 2; ++kc) {
            v16h a0 = load_packed(&sm.u.g.cfP[((2*w)  *2 + kc) * 512], lane);
            v16h a1 = load_packed(&sm.u.g.cfP[((2*w+1)*2 + kc) * 512], lane);
#pragma unroll
            for (int nt = 0; nt < 6; ++nt) {
                v16h bf = load_packed(w1cP + (kc*6 + nt) * 512, lane);
                acc[0][nt] = wmma32(a0, bf, acc[0][nt]);
                acc[1][nt] = wmma32(a1, bf, acc[1][nt]);
            }
        }
#pragma unroll
        for (int mi = 0; mi < 2; ++mi) {
            const int ii = 2*w + mi;
#pragma unroll
            for (int nt = 0; nt < 6; ++nt) {
                const int f = nt*16 + lane15;
                const int kc = f >> 5, kr = f & 31;
                const int h2 = (kr >> 3) & 1;
                const int e  = (kr & 7) | ((kr >> 1) & 8);
#pragma unroll
                for (int r = 0; r < 8; ++r) {
                    const int jj = r + hi*8;
                    float v = acc[mi][nt][r] + sm.hAi[ii][f] + sm.u.g.hBj[jj][f];
                    v = v / (1.f + __expf(-v));          // silu
                    sm.u.g.xP[((ii*3 + kc)*32 + jj + h2*16)*16 + e] = (half_t)v;
                }
            }
        }
        __syncthreads();
        // ---- GEMM2: msgs = silu_x @ W2^T + b2 (kept in registers) ----
#pragma unroll
        for (int mi = 0; mi < 2; ++mi)
#pragma unroll
            for (int nt = 0; nt < 6; ++nt) acc[mi][nt] = vz;
#pragma unroll
        for (int kc = 0; kc < 3; ++kc) {
            v16h a0 = load_packed(&sm.u.g.xP[((2*w)  *3 + kc) * 512], lane);
            v16h a1 = load_packed(&sm.u.g.xP[((2*w+1)*3 + kc) * 512], lane);
#pragma unroll
            for (int nt = 0; nt < 6; ++nt) {
                v16h bf = load_packed(w2P + (kc*6 + nt) * 512, lane);
                acc[0][nt] = wmma32(a0, bf, acc[0][nt]);
                acc[1][nt] = wmma32(a1, bf, acc[1][nt]);
            }
        }
#pragma unroll
        for (int mi = 0; mi < 2; ++mi)
#pragma unroll
            for (int nt = 0; nt < 6; ++nt)
#pragma unroll
                for (int r = 0; r < 8; ++r) acc[mi][nt][r] += sm.b2s[nt*16 + lane15];
        __syncthreads();
        // ---- stage 4 pass 1: online softmax state update, write probs ----
#pragma unroll
        for (int mi = 0; mi < 2; ++mi) {
            const int ii = 2*w + mi;
            const int rdeg = sm.rowdeg[ii];
#pragma unroll
            for (int nt = 0; nt < 6; ++nt) {
                const int f = nt*16 + lane15;
                float z[8]; float mx = -1e30f;
#pragma unroll
                for (int r = 0; r < 8; ++r) {
                    const int jj = r + hi*8;
                    float am = (rdeg == 0) ? 0.f : (sm.adjt[ii][jj] != 0 ? 0.f : -1e30f);
                    z[r] = acc[mi][nt][r] + am;
                    mx = fmaxf(mx, z[r]);
                }
                mx = fmaxf(mx, __shfl_xor(mx, 16, 32));
                float* st = (f < 64) ? sm.cstate[ii][f] : sm.nstate[ii][f - 64];
                const float m_old = st[0];
                const float m_new = fmaxf(m_old, mx);
                const float fac = __expf(m_old - m_new);
                float se = 0.f, se2 = 0.f, e[8];
#pragma unroll
                for (int r = 0; r < 8; ++r) {
                    e[r] = (z[r] < -1e29f) ? 0.f : __expf(z[r] - m_new);
                    se += e[r]; se2 += e[r]*e[r];
                }
                se  += __shfl_xor(se, 16, 32);
                se2 += __shfl_xor(se2, 16, 32);
                if (lane < 16) {
                    st[0] = m_new;
                    st[1] = st[1]*fac + se;
                    st[2] = st[2]*fac*fac + se2;
                    if (f < 64) sm.u.a.cfac[ii][f] = fac; else sm.u.a.nfac[ii][f - 64] = fac;
                }
                if (f >= 64) {
#pragma unroll
                    for (int r = 0; r < 8; ++r) sm.u.a.nodeP[ii][r + hi*8][f - 64] = e[r];
                } else if (f < 32) {
#pragma unroll
                    for (int r = 0; r < 8; ++r) sm.u.a.coordP[ii][r + hi*8][f] = e[r];
                }
            }
        }
        // cooperative loads for accumulation
        for (int idx = tid; idx < 16*DM; idx += 256) {
            int jj = idx / DM, k = idx % DM;
            sm.u.a.qj[jj][k] = (half_t)q_ws[(b*NN + j0 + jj)*DM + k];
        }
        for (int idx = tid; idx < 16*64*3; idx += 256) {
            int jj = idx / 192, r = idx % 192, p = r / 3, d = r % 3;
            sm.u.a.projj[jj][p][d] = proj_ws[((b*3 + d)*NN + j0 + jj)*64 + p];
        }
        __syncthreads();
        // ---- accumulate: node attention (all heads) + coord channels p<32 ----
        for (int idx = tid; idx < 16*32*12; idx += 256) {
            const int ii = idx / 384, r2 = idx % 384, h = r2 / 12, d = r2 % 12;
            const float fac = sm.u.a.nfac[ii][h];
            float s = 0.f;
#pragma unroll
            for (int jj = 0; jj < 16; ++jj)
                s += sm.u.a.nodeP[ii][jj][h] * (float)sm.u.a.qj[jj][h*12 + d];
            sm.nacc[ii][h][d] = sm.nacc[ii][h][d]*fac + s;
        }
        for (int idx = tid; idx < 16*32; idx += 256) {
            const int ii = idx / 32, p = idx % 32;
            const float fac = sm.u.a.cfac[ii][p];
            const float pi0 = sm.proji[ii][p][0], pi1 = sm.proji[ii][p][1], pi2 = sm.proji[ii][p][2];
            float a0 = 0.f, a1 = 0.f, a2 = 0.f;
#pragma unroll
            for (int jj = 0; jj < 16; ++jj) {
                float v0 = pi0 - sm.u.a.projj[jj][p][0];
                float v1 = pi1 - sm.u.a.projj[jj][p][1];
                float v2 = pi2 - sm.u.a.projj[jj][p][2];
                float inv = 1.f / (sqrtf(fmaxf(v0*v0 + v1*v1 + v2*v2, 1e-12f)) + 1e-6f);
                float e = sm.u.a.coordP[ii][jj][p];
                a0 += e*v0*inv; a1 += e*v1*inv; a2 += e*v2*inv;
            }
            sm.cacc[ii][p][0] = sm.cacc[ii][p][0]*fac + a0;
            sm.cacc[ii][p][1] = sm.cacc[ii][p][1]*fac + a1;
            sm.cacc[ii][p][2] = sm.cacc[ii][p][2]*fac + a2;
        }
        __syncthreads();
        // ---- stage 4 pass 2: coord channels 32..63 (probs recomputed) ----
#pragma unroll
        for (int mi = 0; mi < 2; ++mi) {
            const int ii = 2*w + mi;
            const int rdeg = sm.rowdeg[ii];
#pragma unroll
            for (int nt = 2; nt < 4; ++nt) {
                const int f = nt*16 + lane15;
                const float m_now = sm.cstate[ii][f][0];
#pragma unroll
                for (int r = 0; r < 8; ++r) {
                    const int jj = r + hi*8;
                    float am = (rdeg == 0) ? 0.f : (sm.adjt[ii][jj] != 0 ? 0.f : -1e30f);
                    float zz = acc[mi][nt][r] + am;
                    sm.u.a.coordP[ii][jj][f - 32] = (zz < -1e29f) ? 0.f : __expf(zz - m_now);
                }
            }
        }
        __syncthreads();
        for (int idx = tid; idx < 16*32; idx += 256) {
            const int ii = idx / 32, p = 32 + (idx % 32);
            const float fac = sm.u.a.cfac[ii][p];
            const float pi0 = sm.proji[ii][p][0], pi1 = sm.proji[ii][p][1], pi2 = sm.proji[ii][p][2];
            float a0 = 0.f, a1 = 0.f, a2 = 0.f;
#pragma unroll
            for (int jj = 0; jj < 16; ++jj) {
                float v0 = pi0 - sm.u.a.projj[jj][p][0];
                float v1 = pi1 - sm.u.a.projj[jj][p][1];
                float v2 = pi2 - sm.u.a.projj[jj][p][2];
                float inv = 1.f / (sqrtf(fmaxf(v0*v0 + v1*v1 + v2*v2, 1e-12f)) + 1e-6f);
                float e = sm.u.a.coordP[ii][jj][p - 32];
                a0 += e*v0*inv; a1 += e*v1*inv; a2 += e*v2*inv;
            }
            sm.cacc[ii][p][0] = sm.cacc[ii][p][0]*fac + a0;
            sm.cacc[ii][p][1] = sm.cacc[ii][p][1]*fac + a1;
            sm.cacc[ii][p][2] = sm.cacc[ii][p][2]*fac + a2;
        }
        __syncthreads();
    }

    // ---- finalize: Out (f16, for out_w GEMM) and coord update vectors ----
    for (int idx = tid; idx < 16*32*12; idx += 256) {
        const int ii = idx / 384, r2 = idx % 384, h = r2 / 12, d = r2 % 12;
        const float l = sm.nstate[ii][h][1], ssq = sm.nstate[ii][h][2];
        const float v = sm.nacc[ii][h][d] * sqrtf(ssq) / (l*l);
        out_h[(b*NN + i0 + ii)*DM + h*12 + d] = (half_t)v;
    }
    for (int idx = tid; idx < 16*64*3; idx += 256) {
        const int ii = idx / 192, r2 = idx % 192, p = r2 / 3, d = r2 % 3;
        const float l = sm.cstate[ii][p][1], ssq = sm.cstate[ii][p][2];
        upd_ws[((b*3 + d)*NN + i0 + ii)*64 + p] = sm.cacc[ii][p][d] * sqrtf(ssq) / (l*l);
    }
}

// ---------------------------------------------------------------------------
// K5: coord_out[b,s,n,d] = coords + sum_p upd[b,d,n,p] * ca_attn_w[s,p]
// ---------------------------------------------------------------------------
__global__ void k_coord_out(const float* __restrict__ coords, const float* __restrict__ upd,
                            const float* __restrict__ aw, float* __restrict__ out) {
    const int bn = blockIdx.x;  const int b = bn >> 8, n = bn & 255;
    __shared__ float W[64][64];
    __shared__ float ul[3][64];
    const int t = threadIdx.x;  // 192
    for (int i = t; i < 4096; i += 192) W[i >> 6][i & 63] = aw[i];
    { int d = t / 64, p = t % 64; ul[d][p] = upd[((b*3 + d)*NN + n)*64 + p]; }
    __syncthreads();
    const int s = t / 3, d = t % 3;
    float acc = 0.f;
#pragma unroll 8
    for (int p = 0; p < 64; ++p) acc += ul[d][p] * W[s][p];
    const int idx = ((b*SS + s)*NN + n)*3 + d;
    out[idx] = coords[idx] + acc;
}

// ---------------------------------------------------------------------------
// Launch
// ---------------------------------------------------------------------------
extern "C" void kernel_launch(void* const* d_in, const int* in_sizes, int n_in,
                              void* d_out, int out_size, void* d_ws, size_t ws_size,
                              hipStream_t stream) {
    (void)in_sizes; (void)n_in; (void)out_size; (void)ws_size;
    const float* coords     = (const float*)d_in[0];
    const float* node_feats = (const float*)d_in[1];
    const int*   adj        = (const int*)d_in[2];
    const float* mask       = (const float*)d_in[3];
    const float* em_set_w   = (const float*)d_in[4];
    const float* em_ln_g    = (const float*)d_in[5];
    const float* em_ln_b    = (const float*)d_in[6];
    const float* np_w       = (const float*)d_in[7];
    const float* np_b       = (const float*)d_in[8];
    const float* w1         = (const float*)d_in[9];
    const float* b1         = (const float*)d_in[10];
    const float* w2         = (const float*)d_in[11];
    const float* b2         = (const float*)d_in[12];
    const float* na_ln_g    = (const float*)d_in[13];
    const float* na_ln_b    = (const float*)d_in[14];
    const float* in_w       = (const float*)d_in[15];
    const float* in_b       = (const float*)d_in[16];
    const float* out_w      = (const float*)d_in[17];
    const float* out_b      = (const float*)d_in[18];
    const float* ca_set_w   = (const float*)d_in[19];
    const float* ca_proj_w  = (const float*)d_in[20];
    const float* ca_attn_w  = (const float*)d_in[21];

    char* ws = (char*)d_ws;
    float*  c_p    = (float*)(ws + OFF_C);
    float*  cn_p   = (float*)(ws + OFF_CN);
    float*  proj_p = (float*)(ws + OFF_PROJ);
    float*  hA_p   = (float*)(ws + OFF_HA);
    float*  hB_p   = (float*)(ws + OFF_HB);
    half_t* LNa_p  = (half_t*)(ws + OFF_LNA);
    float*  q_p    = (float*)(ws + OFF_Q);
    half_t* outh_p = (half_t*)(ws + OFF_OUTH);
    float*  upd_p  = (float*)(ws + OFF_UPD);
    int*    rd_p   = (int*)(ws + OFF_RD);
    half_t* w1cP_p = (half_t*)(ws + OFF_W1C);
    half_t* w2P_p  = (half_t*)(ws + OFF_W2T);
    half_t* inP_p  = (half_t*)(ws + OFF_INT);
    half_t* outP_p = (half_t*)(ws + OFF_OUTT);

    float* coord_out = (float*)d_out;
    float* node_out  = coord_out + (size_t)BB*SS*NN*3;

    k_prep_weights<<<256, 256, 0, stream>>>(w1, w2, in_w, out_w, w1cP_p, w2P_p, inP_p, outP_p);
    k_coord_prep<<<BB*SS, 256, 0, stream>>>(coords, mask, em_set_w, ca_set_w, c_p, cn_p);
    k_proj<<<BB*NN, 192, 0, stream>>>(cn_p, ca_proj_w, proj_p);
    k_node_prep<<<BB*NN, 128, 0, stream>>>(node_feats, em_ln_g, em_ln_b, na_ln_g, na_ln_b,
                                           np_w, np_b, w1, b1, adj, LNa_p, hA_p, hB_p, rd_p);
    k_gemm<<<dim3((BB*NN)/128, DM/64), 256, 0, stream>>>(LNa_p, inP_p, in_b, nullptr, q_p,
                                                         BB*NN, DM, DM);
    k_fused<<<BB*16, 256, 0, stream>>>(c_p, proj_p, hA_p, hB_p, q_p, rd_p, adj,
                                       w1cP_p, w2P_p, b2, outh_p, upd_p);
    k_gemm<<<dim3((BB*NN)/128, DM/64), 256, 0, stream>>>(outh_p, outP_p, out_b, node_feats,
                                                         node_out, BB*NN, DM, DM);
    k_coord_out<<<BB*NN, 192, 0, stream>>>(coords, upd_p, ca_attn_w, coord_out);
}
